// STN_2199023256248
// MI455X (gfx1250) — matrix operand order repaired and checked by reading, not compiled, not run on hardware
//
#include <hip/hip_runtime.h>
#include <hip/hip_bf16.h>
#include <math.h>

typedef __attribute__((ext_vector_type(16))) _Float16 v16h;
typedef __attribute__((ext_vector_type(8)))  float    v8f;

#define BATCH 128
#define H_ 224
#define W_ 224

__device__ __forceinline__ v8f wmma16x16x32(v16h a, v16h b, v8f c) {
  return __builtin_amdgcn_wmma_f32_16x16x32_f16(
      /*neg_a=*/false, a, /*neg_b=*/false, b,
      /*c_mod=*/(short)0, c, /*reuse_a=*/false, /*reuse_b=*/false);
}

// CDNA5 ISA 7.12.2: A (16x32 f16) layout. half = lane>>4, mrow = lane&15.
__device__ __forceinline__ int a_kidx(int v, int half) {
  return ((v >> 2) << 4) + half * 8 + ((v & 3) << 1);
}
// B (32x16 f16): n = lane&15, halfK = lane>>4.
__device__ __forceinline__ int b_kidx(int v, int halfK) {
  return ((v >> 2) << 4) + halfK * 8 + ((v & 3) << 1);
}

// async global->LDS copy helpers (CDNA5 §10.7/§15.18.3; ASYNCcnt-tracked)
__device__ __forceinline__ void async_g2l_b32(unsigned ldsoff, const void* gp) {
  asm volatile("global_load_async_to_lds_b32 %0, %1, off"
               :: "v"(ldsoff), "v"((unsigned long long)(uintptr_t)gp)
               : "memory");
}
__device__ __forceinline__ void async_g2l_b128(unsigned ldsoff, const void* gp) {
  asm volatile("global_load_async_to_lds_b128 %0, %1, off"
               :: "v"(ldsoff), "v"((unsigned long long)(uintptr_t)gp)
               : "memory");
}
__device__ __forceinline__ void wait_async0() {
  asm volatile("s_wait_asynccnt 0x0" ::: "memory");
}

// ---------------- conv1 (3x3, 3->8) + ReLU + 2x2 maxpool ------------------
// Block (224 thr = 7 waves) owns (b, yp): conv rows 2yp,2yp+1.
// LDS: input rows 2yp-1..2yp+2, x in [-1,224], 3ch -> [4][226*3] f32 (off 0).
#define C1W 678   // 226*3

__device__ __forceinline__ v16h a_from_lds_c1(const float* sm, int rbase,
                                              int xl, int half) {
  v16h a;
#pragma unroll
  for (int v = 0; v < 8; ++v) {
#pragma unroll
    for (int e = 0; e < 2; ++e) {
      int k = a_kidx(v, half) + e;
      float val = 0.f;
      if (k < 27) {
        int ky = k / 9, r = k % 9, kx = r / 3, ci = r % 3;
        val = sm[(rbase + ky) * C1W + (xl + kx) * 3 + ci];
      }
      a[2 * v + e] = (_Float16)val;
    }
  }
  return a;
}

__global__ __launch_bounds__(224)
void conv1_pool(const float* __restrict__ xin, const float* __restrict__ w,
                const float* __restrict__ bias, _Float16* __restrict__ t1) {
  __shared__ __align__(16) float sm[4 * C1W];
  const int b  = blockIdx.x / 112;
  const int yp = blockIdx.x % 112;

  // interior: async DMA copy of 4 input rows (672 f32 each) into LDS
  for (int q = threadIdx.x; q < 4 * 672; q += 224) {       // 12 per thread
    int r = q / 672, c = q % 672;                           // c = ix*3+ci
    int iy = 2 * yp - 1 + r;
    unsigned ldsoff = (unsigned)((r * C1W + 3 + c) * 4);
    if (iy >= 0 && iy < H_)
      async_g2l_b32(ldsoff, xin + ((size_t)(b * H_ + iy) * 672 + c));
    else
      sm[r * C1W + 3 + c] = 0.f;
  }
  // zero border columns (x = -1 and x = 224)
  if (threadIdx.x < 24) {
    int r = threadIdx.x / 6, p = threadIdx.x % 6;
    int c = (p < 3) ? p : (675 + p - 3);
    sm[r * C1W + c] = 0.f;
  }

  const int lane = threadIdx.x & 31;
  const int wv   = threadIdx.x >> 5;            // 0..6
  const int half = lane >> 4, mrow = lane & 15;
  const int n = lane & 15, kh = lane >> 4;

  // B fragment while DMA is in flight: [27 x 8] padded to [32 x 16]
  v16h bf;
#pragma unroll
  for (int v = 0; v < 8; ++v) {
#pragma unroll
    for (int e = 0; e < 2; ++e) {
      int k = b_kidx(v, kh) + e;
      float val = (k < 27 && n < 8) ? w[k * 8 + n] : 0.f;
      bf[2 * v + e] = (_Float16)val;
    }
  }
  const float bv = (n < 8) ? bias[n] : 0.f;

  wait_async0();
  __syncthreads();

#pragma unroll
  for (int t = 0; t < 2; ++t) {
    const int tile = wv * 2 + t;                // 0..13
    const int x0 = tile * 16;
    v16h a0 = a_from_lds_c1(sm, 0, x0 + mrow, half);  // conv row 2yp
    v16h a1 = a_from_lds_c1(sm, 1, x0 + mrow, half);  // conv row 2yp+1
    v8f zero = {};
    v8f d0 = wmma16x16x32(a0, bf, zero);
    v8f d1 = wmma16x16x32(a1, bf, zero);
#pragma unroll
    for (int j = 0; j < 4; ++j) {
      float mx = fmaxf(fmaxf(d0[2 * j], d0[2 * j + 1]),
                       fmaxf(d1[2 * j], d1[2 * j + 1]));
      float r = fmaxf(mx + bv, 0.f);
      int px = tile * 8 + half * 4 + j;
      if (n < 8)
        t1[((size_t)(b * 112 + yp) * 112 + px) * 8 + n] = (_Float16)r;
    }
  }
}

// ---------------- conv2 (3x3, 8->10) + ReLU + 2x2 maxpool -----------------
// Block (224 thr = 7 waves) owns (b, yp@56). LDS rows 2yp-1..2yp+2 of t1.
#define C2W 912   // 114*8

__device__ __forceinline__ v16h a_from_lds_c2(const _Float16* sm, int rbase,
                                              int xl, int half, int s) {
  v16h a;
#pragma unroll
  for (int v = 0; v < 8; ++v) {
#pragma unroll
    for (int e = 0; e < 2; ++e) {
      int k = s * 32 + a_kidx(v, half) + e;
      _Float16 val = (_Float16)0.f;
      if (k < 72) {
        int ky = k / 24, r = k % 24, kx = r / 8, ci = r & 7;
        val = sm[(rbase + ky) * C2W + (xl + kx) * 8 + ci];
      }
      a[2 * v + e] = val;
    }
  }
  return a;
}

__global__ __launch_bounds__(224)
void conv2_pool(const _Float16* __restrict__ t1, const float* __restrict__ w,
                const float* __restrict__ bias, _Float16* __restrict__ t2) {
  __shared__ __align__(16) _Float16 sm[4 * C2W];
  const int b  = blockIdx.x / 56;
  const int yp = blockIdx.x % 56;

  // interior: async DMA, 4 rows x 112 chunks of 16B (16B aligned both sides)
  for (int q = threadIdx.x; q < 4 * 112; q += 224) {        // 2 per thread
    int r = q / 112, chunk = q % 112;
    int iy = 2 * yp - 1 + r;
    unsigned ldsoff = (unsigned)(r * C2W * 2 + 16 + chunk * 16);
    if (iy >= 0 && iy < 112)
      async_g2l_b128(ldsoff, t1 + ((size_t)(b * 112 + iy) * 896 + chunk * 8));
    else {
#pragma unroll
      for (int z = 0; z < 8; ++z)
        sm[r * C2W + 8 + chunk * 8 + z] = (_Float16)0.f;
    }
  }
  // zero border columns (ix = -1 and ix = 112)
  if (threadIdx.x < 64) {
    int r = threadIdx.x / 16, p = threadIdx.x % 16;
    int c = (p < 8) ? p : (904 + p - 8);
    sm[r * C2W + c] = (_Float16)0.f;
  }

  const int lane = threadIdx.x & 31;
  const int tile = threadIdx.x >> 5;            // 0..6, one x-tile per wave
  const int half = lane >> 4, mrow = lane & 15;
  const int n = lane & 15, kh = lane >> 4;

  // B fragments while DMA in flight: [72 x 10] pad [96 x 16]
  v16h bf[3];
#pragma unroll
  for (int s = 0; s < 3; ++s)
#pragma unroll
    for (int v = 0; v < 8; ++v)
#pragma unroll
      for (int e = 0; e < 2; ++e) {
        int k = s * 32 + b_kidx(v, kh) + e;
        float val = (k < 72 && n < 10) ? w[k * 10 + n] : 0.f;
        bf[s][2 * v + e] = (_Float16)val;
      }

  wait_async0();
  __syncthreads();

  const int x0 = tile * 16;
  v8f d0 = {}, d1 = {};
#pragma unroll
  for (int s = 0; s < 3; ++s) {
    v16h a0 = a_from_lds_c2(sm, 0, x0 + mrow, half, s);
    v16h a1 = a_from_lds_c2(sm, 1, x0 + mrow, half, s);
    d0 = wmma16x16x32(a0, bf[s], d0);
    d1 = wmma16x16x32(a1, bf[s], d1);
  }

  const float bv = (n < 10) ? bias[n] : 0.f;
#pragma unroll
  for (int j = 0; j < 4; ++j) {
    float mx = fmaxf(fmaxf(d0[2 * j], d0[2 * j + 1]),
                     fmaxf(d1[2 * j], d1[2 * j + 1]));
    float r = fmaxf(mx + bv, 0.f);
    int px = tile * 8 + half * 4 + j;
    if (n < 10)
      t2[((size_t)(b * 56 + yp) * 56 + px) * 10 + n] = (_Float16)r;
  }
}

// ---------------- fc1: [128,31360] x [31360,32], deterministic split-K ----
__global__ __launch_bounds__(256)
void fc1_splitk(const _Float16* __restrict__ t2, const float* __restrict__ w,
                float* __restrict__ t3p) {
  const int lane = threadIdx.x & 31;
  const int wave = blockIdx.x * 8 + (threadIdx.x >> 5);   // 320 waves
  const int chunk = wave % 20;
  const int ntile = (wave / 20) & 1;
  const int mtile = wave / 40;
  const int half = lane >> 4, mrow = lane & 15;
  const int n = lane & 15, kh = lane >> 4;
  const int mg = mtile * 16 + mrow;
  const int ng = ntile * 16 + n;

  const _Float16* arow = t2 + (size_t)mg * 31360;
  v8f acc = {};
  for (int t = 0; t < 49; ++t) {
    int kb = (chunk * 49 + t) * 32;
    v16h a, bfr;
#pragma unroll
    for (int v = 0; v < 8; ++v) {
      int ka = kb + a_kidx(v, half);
      a[2 * v]     = arow[ka];
      a[2 * v + 1] = arow[ka + 1];
      int kbi = kb + b_kidx(v, kh);
      bfr[2 * v]     = (_Float16)w[(size_t)kbi * 32 + ng];
      bfr[2 * v + 1] = (_Float16)w[(size_t)(kbi + 1) * 32 + ng];
    }
    acc = wmma16x16x32(a, bfr, acc);
  }
#pragma unroll
  for (int v = 0; v < 8; ++v) {
    int m = mtile * 16 + v + half * 8;
    t3p[(size_t)chunk * 4096 + m * 32 + ng] = acc[v];
  }
}

// ---------------- reduce partials + bias/relu + fc2 -> theta [128,6] ------
__global__ void fc2_theta(const float* __restrict__ t3p,
                          const float* __restrict__ fc1_b,
                          const float* __restrict__ fc2_w,
                          const float* __restrict__ fc2_b,
                          float* __restrict__ theta) {
  int b = blockIdx.x * blockDim.x + threadIdx.x;
  if (b >= BATCH) return;
  float acc[6];
#pragma unroll
  for (int i = 0; i < 6; ++i) acc[i] = fc2_b[i];
  for (int j = 0; j < 32; ++j) {
    float s = 0.f;
    for (int ch = 0; ch < 20; ++ch) s += t3p[(size_t)ch * 4096 + b * 32 + j];
    float t = fmaxf(s + fc1_b[j], 0.f);
#pragma unroll
    for (int i = 0; i < 6; ++i) acc[i] += t * fc2_w[j * 6 + i];
  }
#pragma unroll
  for (int i = 0; i < 6; ++i) theta[b * 6 + i] = acc[i];
}

// ---------------- affine grid + bilinear sampling -------------------------
__global__ __launch_bounds__(256)
void sampler(const float* __restrict__ xin, const float* __restrict__ theta,
             float* __restrict__ out) {
  int idx = blockIdx.x * 256 + threadIdx.x;           // b*H*W + y*W + x
  if (idx >= BATCH * H_ * W_) return;
  int xc = idx % W_;
  int y  = (idx / W_) % H_;
  int b  = idx / (H_ * W_);
  const float* th = theta + b * 6;
  float xt = -1.f + 2.f * (float)xc / (float)(W_ - 1);
  float yt = -1.f + 2.f * (float)y  / (float)(H_ - 1);
  float gx = th[0] * xt + th[1] * yt + th[2];
  float gy = th[3] * xt + th[4] * yt + th[5];
  float x_s = (gx + 1.f) * 0.5f * (float)(W_ - 1);
  float y_s = (gy + 1.f) * 0.5f * (float)(H_ - 1);
  int x0i = min(max((int)floorf(x_s), 0), W_ - 1);
  int x1i = min(x0i + 1, W_ - 1);
  int y0i = min(max((int)floorf(y_s), 0), H_ - 1);
  int y1i = min(y0i + 1, H_ - 1);
  float wa = ((float)x1i - x_s) * ((float)y1i - y_s);
  float wb = ((float)x1i - x_s) * (y_s - (float)y0i);
  float wc = (x_s - (float)x0i) * ((float)y1i - y_s);
  float wd = (x_s - (float)x0i) * (y_s - (float)y0i);
  const float* base = xin + (size_t)b * H_ * W_ * 3;
  const float* p00 = base + ((size_t)y0i * W_ + x0i) * 3;
  const float* p01 = base + ((size_t)y1i * W_ + x0i) * 3;
  const float* p10 = base + ((size_t)y0i * W_ + x1i) * 3;
  const float* p11 = base + ((size_t)y1i * W_ + x1i) * 3;
  float* o = out + (size_t)idx * 3;
#pragma unroll
  for (int c = 0; c < 3; ++c)
    o[c] = wa * p00[c] + wb * p01[c] + wc * p10[c] + wd * p11[c];
}

extern "C" void kernel_launch(void* const* d_in, const int* in_sizes, int n_in,
                              void* d_out, int out_size, void* d_ws, size_t ws_size,
                              hipStream_t stream) {
  (void)in_sizes; (void)n_in; (void)out_size; (void)ws_size;
  const float* x       = (const float*)d_in[0];
  const float* conv1_w = (const float*)d_in[1];
  const float* conv1_b = (const float*)d_in[2];
  const float* conv2_w = (const float*)d_in[3];
  const float* conv2_b = (const float*)d_in[4];
  const float* fc1_w   = (const float*)d_in[5];
  const float* fc1_b   = (const float*)d_in[6];
  const float* fc2_w   = (const float*)d_in[7];
  const float* fc2_b   = (const float*)d_in[8];
  float* out = (float*)d_out;

  char* ws = (char*)d_ws;
  _Float16* t1   = (_Float16*)(ws);                        // 25,690,112 B
  _Float16* t2   = (_Float16*)(ws + 25690112);             //  8,028,160 B
  float*    t3p  = (float*)   (ws + 25690112 + 8028160);   //    327,680 B
  float*    thet = (float*)   (ws + 25690112 + 8028160 + 327680);

  conv1_pool<<<128 * 112, 224, 0, stream>>>(x, conv1_w, conv1_b, t1);
  conv2_pool<<<128 * 56, 224, 0, stream>>>(t1, conv2_w, conv2_b, t2);
  fc1_splitk<<<40, 256, 0, stream>>>(t2, fc1_w, t3p);
  fc2_theta<<<1, 128, 0, stream>>>(t3p, fc1_b, fc2_w, fc2_b, thet);
  sampler<<<25088, 256, 0, stream>>>(x, thet, out);
}
